// AllusionBERTCRF_67001489818176
// MI455X (gfx1250) — compile-verified, matrix operand order checked by reading, not run on hardware
//
#include <hip/hip_runtime.h>

// ---------------- problem constants ----------------
static constexpr int BB   = 64;
static constexpr int SS   = 512;
static constexpr int MM   = BB * SS;      // 32768 token rows
static constexpr int DBERT = 768;
static constexpr int DDICT = 256;
static constexpr int DIN0  = DBERT + DDICT; // 1024
static constexpr int HH    = 256;
static constexpr int G4H   = 4 * HH;      // 1024 gates per dir
static constexpr int NGATE = 2 * G4H;     // 2048 (both dirs stacked)

typedef __attribute__((ext_vector_type(16))) __bf16 v16bf;
typedef __attribute__((ext_vector_type(8)))  __bf16 v8bf;
typedef __attribute__((ext_vector_type(8)))  float  v8f;

// ---------------- fragment loader ----------------
// For D = A * W^T with A (MxK) and W (NxK) both row-major bf16, the 16x32
// A-fragment and 32x16 B-fragment have IDENTICAL per-lane layouts:
// lane<16: row = lane, K = {k0..k0+7, k0+16..k0+23}
// lane>=16: row = lane-16, K shifted by 8.  Two b128 loads per fragment.
__device__ inline v16bf load_frag(const __bf16* p, int ld) {
  int lane = threadIdx.x & 31;
  const __bf16* q = p + (size_t)(lane & 15) * ld + ((lane >> 4) << 3);
  v8bf lo = *(const v8bf*)(q);
  v8bf hi = *(const v8bf*)(q + 16);
  v16bf f;
#pragma unroll
  for (int i = 0; i < 8; ++i) { f[i] = lo[i]; f[i + 8] = hi[i]; }
  return f;
}

__device__ inline v8f wmma_bf16(v16bf a, v16bf b, v8f c) {
  return __builtin_amdgcn_wmma_f32_16x16x32_bf16(false, a, false, b,
                                                 (short)0, c, false, false);
}

// ---------------- elementwise converts ----------------
__global__ void f32_to_bf16_k(const float* __restrict__ src,
                              __bf16* __restrict__ dst, int n) {
  int i = blockIdx.x * blockDim.x + threadIdx.x;
  if (i < n) dst[i] = (__bf16)src[i];
}

// sequence_output (M,768) f32 -> combined bf16 (M,1024) cols [0,768)
__global__ void pack_seq_k(const float* __restrict__ seq,
                           __bf16* __restrict__ comb, int n) {
  int i = blockIdx.x * blockDim.x + threadIdx.x;
  if (i >= n) return;
  int r = i / DBERT, c = i - r * DBERT;
  comb[(size_t)r * DIN0 + c] = (__bf16)seq[i];
}

// dict gather: summed[r][a] = sum_k emb[idx[r][k]][a] * val[r][k]  (bf16 out)
__global__ void dict_gather_k(const int* __restrict__ idx,
                              const float* __restrict__ val,
                              const float* __restrict__ emb,
                              __bf16* __restrict__ summed) {
  int t = blockIdx.x * blockDim.x + threadIdx.x;   // M * 64 threads
  int row = t >> 6;
  if (row >= MM) return;
  int c4 = (t & 63) << 2;
  const int*   ip = idx + row * 5;
  const float* vp = val + row * 5;
  float acc0 = 0.f, acc1 = 0.f, acc2 = 0.f, acc3 = 0.f;
#pragma unroll
  for (int k = 0; k < 5; ++k) {
    const float4 e = *(const float4*)(emb + (size_t)ip[k] * DDICT + c4);
    float v = vp[k];
    acc0 += e.x * v; acc1 += e.y * v; acc2 += e.z * v; acc3 += e.w * v;
  }
  __bf16* o = summed + (size_t)row * DDICT + c4;
  o[0] = (__bf16)acc0; o[1] = (__bf16)acc1;
  o[2] = (__bf16)acc2; o[3] = (__bf16)acc3;
}

// ---------------- WMMA GEMM: C = A * W^T (+bias)(relu)(f32|bf16 out) -------
// block = 256 thr (8 waves); block tile 64M x 256N; wave tile 32M x 64N.
// grid = (M/64, N/256). K % 32 == 0.
#define GF_RELU 1
#define GF_BF16 2
__global__ __launch_bounds__(256) void gemm_wmma_k(
    const __bf16* __restrict__ A, int lda,
    const __bf16* __restrict__ W, int ldw,
    const float* __restrict__ bias,
    float* __restrict__ Cf, __bf16* __restrict__ Cb, int ldc,
    int K, int flags) {
  int wave = threadIdx.x >> 5;
  int m0 = blockIdx.x * 64  + (wave & 1) * 32;
  int n0 = blockIdx.y * 256 + (wave >> 1) * 64;
  v8f acc[2][4];
#pragma unroll
  for (int mi = 0; mi < 2; ++mi)
#pragma unroll
    for (int n = 0; n < 4; ++n)
#pragma unroll
      for (int i = 0; i < 8; ++i) acc[mi][n][i] = 0.f;

  for (int k0 = 0; k0 < K; k0 += 32) {
    v16bf a0 = load_frag(A + (size_t)m0 * lda + k0, lda);
    v16bf a1 = load_frag(A + (size_t)(m0 + 16) * lda + k0, lda);
#pragma unroll
    for (int n = 0; n < 4; ++n) {
      v16bf bf = load_frag(W + (size_t)(n0 + n * 16) * ldw + k0, ldw);
      acc[0][n] = wmma_bf16(a0, bf, acc[0][n]);
      acc[1][n] = wmma_bf16(a1, bf, acc[1][n]);
    }
  }
  int lane  = threadIdx.x & 31;
  int rbase = (lane >> 4) << 3;   // 0 or 8
  int col_l = lane & 15;
#pragma unroll
  for (int mi = 0; mi < 2; ++mi)
#pragma unroll
    for (int n = 0; n < 4; ++n) {
      int col = n0 + n * 16 + col_l;
      float bv = bias ? bias[col] : 0.f;
#pragma unroll
      for (int i = 0; i < 8; ++i) {
        int row = m0 + mi * 16 + rbase + i;
        float v = acc[mi][n][i] + bv;
        if (flags & GF_RELU) v = fmaxf(v, 0.f);
        if (flags & GF_BF16) Cb[(size_t)row * ldc + col] = (__bf16)v;
        else                 Cf[(size_t)row * ldc + col] = v;
      }
    }
}

// ---------------- persistent BiLSTM recurrence -----------------------------
// One workgroup per direction (grid.x = 2), 1024 threads = 32 waves.
// LDS: h_bf16 (64x256, 32KB) + gates f32 (64x1024, 256KB) = 288KB.
// Per step: G = h * Whh_d^T via WMMA (4 Mtiles x 64 Ntiles, 8 tiles/wave),
// then fused cell update (c kept in registers, 16 elems/thread).
__global__ __launch_bounds__(1024) void lstm_rec_k(
    const __bf16* __restrict__ Whh,   // (2, 1024, 256) bf16
    const float*  __restrict__ pre,   // (M, 2048) f32, bias pre-added
    __bf16* __restrict__ hout) {      // (M, 512) bf16; this dir writes 256 cols
  int dir = blockIdx.x;
  extern __shared__ char smem[];
  __bf16* h_bf  = (__bf16*)smem;                        // 64*256
  float*  gates = (float*)(smem + BB * HH * sizeof(__bf16)); // 64*1024
  const __bf16* Wd = Whh + (size_t)dir * G4H * HH;
  int tid = threadIdx.x;
  for (int i = tid; i < BB * HH; i += 1024) h_bf[i] = (__bf16)0.0f;
  float c[16];
#pragma unroll
  for (int j = 0; j < 16; ++j) c[j] = 0.f;

  int wave  = tid >> 5;
  int mt    = wave >> 3;          // 0..3  (batch tile of 16)
  int nt0   = (wave & 7) * 8;     // 8 N-tiles of 16 per wave
  int lane  = tid & 31;
  int rbase = (lane >> 4) << 3;
  int col_l = lane & 15;
  __syncthreads();

  for (int s = 0; s < SS; ++s) {
    int t = dir ? (SS - 1 - s) : s;
    v8f acc[8];
#pragma unroll
    for (int n = 0; n < 8; ++n)
#pragma unroll
      for (int i = 0; i < 8; ++i) acc[n][i] = 0.f;
    for (int k0 = 0; k0 < HH; k0 += 32) {
      v16bf a = load_frag(h_bf + (size_t)(mt * 16) * HH + k0, HH);
#pragma unroll
      for (int n = 0; n < 8; ++n) {
        v16bf bf = load_frag(Wd + (size_t)((nt0 + n) * 16) * HH + k0, HH);
        acc[n] = wmma_bf16(a, bf, acc[n]);
      }
    }
#pragma unroll
    for (int n = 0; n < 8; ++n) {
      int col = (nt0 + n) * 16 + col_l;
#pragma unroll
      for (int i = 0; i < 8; ++i)
        gates[(mt * 16 + rbase + i) * G4H + col] = acc[n][i];
    }
    __syncthreads();
    // cell update: element e = tid + j*1024 of the 64x256 state
#pragma unroll
    for (int j = 0; j < 16; ++j) {
      int e  = tid + j * 1024;
      int bb = e >> 8, hh = e & 255;
      const float* pr = pre + (size_t)(bb * SS + t) * NGATE + dir * G4H;
      float gi = gates[bb * G4H + hh]            + pr[hh];
      float gf = gates[bb * G4H + HH + hh]       + pr[HH + hh];
      float gg = gates[bb * G4H + 2 * HH + hh]   + pr[2 * HH + hh];
      float go = gates[bb * G4H + 3 * HH + hh]   + pr[3 * HH + hh];
      float si = 1.f / (1.f + __expf(-gi));
      float sf = 1.f / (1.f + __expf(-gf));
      float so = 1.f / (1.f + __expf(-go));
      float cn = sf * c[j] + si * tanhf(gg);
      c[j] = cn;
      float hv = so * tanhf(cn);
      h_bf[bb * HH + hh] = (__bf16)hv;
      hout[(size_t)(bb * SS + t) * (2 * HH) + dir * HH + hh] = (__bf16)hv;
    }
    __syncthreads();
  }
}

// ---------------- emissions: (M,512) bf16 @ pos_W^T (3,512) + pos_b --------
__global__ void emissions_k(const __bf16* __restrict__ h,
                            const float* __restrict__ pW,
                            const float* __restrict__ pb,
                            float* __restrict__ em) {
  int r = blockIdx.x * blockDim.x + threadIdx.x;
  if (r >= MM) return;
  const __bf16* hr = h + (size_t)r * (2 * HH);
  float a0 = pb[0], a1 = pb[1], a2 = pb[2];
  for (int k = 0; k < 2 * HH; ++k) {
    float hv = (float)hr[k];
    a0 += hv * pW[k];
    a1 += hv * pW[512 + k];
    a2 += hv * pW[1024 + k];
  }
  em[r * 3 + 0] = a0; em[r * 3 + 1] = a1; em[r * 3 + 2] = a2;
}

// ---------------- CRF log-likelihood + weighted mean loss ------------------
__device__ inline float lse3(float a, float b, float c) {
  float m = fmaxf(a, fmaxf(b, c));
  return m + __logf(__expf(a - m) + __expf(b - m) + __expf(c - m));
}

__global__ void crf_k(const float* __restrict__ em,
                      const int* __restrict__ tags,
                      const int* __restrict__ mask,
                      const float* __restrict__ start,
                      const float* __restrict__ endv,
                      const float* __restrict__ trans,
                      float* __restrict__ out) {
  __shared__ float red[BB];
  int b = threadIdx.x;            // 64 threads, one per sequence
  float tr[9];
#pragma unroll
  for (int i = 0; i < 9; ++i) tr[i] = trans[i];
  const float* eb = em + (size_t)b * SS * 3;
  const int*   tb = tags + b * SS;
  const int*   mb = mask + b * SS;
  int t0 = tb[0];
  float num = start[t0] + eb[t0];
  int prev = t0;
  float al0 = start[0] + eb[0];
  float al1 = start[1] + eb[1];
  float al2 = start[2] + eb[2];
  float wsum = (t0 > 0) ? 150.f : 1.f;
  for (int s = 1; s < SS; ++s) {
    float e0 = eb[s * 3], e1 = eb[s * 3 + 1], e2 = eb[s * 3 + 2];
    int   tt = tb[s];
    float mt = (mb[s] > 0) ? 1.f : 0.f;
    float ec = (tt == 0) ? e0 : ((tt == 1) ? e1 : e2);
    num += (tr[prev * 3 + tt] + ec) * mt;
    if (mt > 0.f) prev = tt;
    float n0 = lse3(al0 + tr[0], al1 + tr[3], al2 + tr[6]) + e0;
    float n1 = lse3(al0 + tr[1], al1 + tr[4], al2 + tr[7]) + e1;
    float n2 = lse3(al0 + tr[2], al1 + tr[5], al2 + tr[8]) + e2;
    if (mt > 0.f) { al0 = n0; al1 = n1; al2 = n2; }
    wsum += (tt > 0) ? 150.f : 1.f;
  }
  num += endv[prev];
  float logZ = lse3(al0 + endv[0], al1 + endv[1], al2 + endv[2]);
  float llh = num - logZ;
  red[b] = -llh * (wsum / (float)SS);
  __syncthreads();
  if (b == 0) {
    float sm = 0.f;
    for (int i = 0; i < BB; ++i) sm += red[i];
    out[0] = sm / (float)BB;
  }
}

// ---------------- host orchestration ----------------
extern "C" void kernel_launch(void* const* d_in, const int* in_sizes, int n_in,
                              void* d_out, int out_size, void* d_ws, size_t ws_size,
                              hipStream_t stream) {
  const float* seq     = (const float*)d_in[0];
  const int*   didx    = (const int*)  d_in[1];
  const float* dval    = (const float*)d_in[2];
  const int*   amask   = (const int*)  d_in[3];
  const int*   plabel  = (const int*)  d_in[4];
  const float* demb    = (const float*)d_in[5];
  const float* dW      = (const float*)d_in[6];
  const float* db      = (const float*)d_in[7];
  const float* l0Wih   = (const float*)d_in[8];
  const float* l0Whh   = (const float*)d_in[9];
  const float* l0b     = (const float*)d_in[10];
  const float* l1Wih   = (const float*)d_in[11];
  const float* l1Whh   = (const float*)d_in[12];
  const float* l1b     = (const float*)d_in[13];
  const float* posW    = (const float*)d_in[14];
  const float* posb    = (const float*)d_in[15];
  const float* crfS    = (const float*)d_in[16];
  const float* crfE    = (const float*)d_in[17];
  const float* crfT    = (const float*)d_in[18];

  char* ws = (char*)d_ws;
  size_t off = 0;
  auto take = [&](size_t bytes) { char* p = ws + off; off += (bytes + 255) & ~size_t(255); return p; };
  __bf16* combined = (__bf16*)take((size_t)MM * DIN0 * 2);   // 64 MiB
  float*  pre      = (float*) take((size_t)MM * NGATE * 4);  // 256 MiB (reused L0/L1)
  __bf16* h0       = (__bf16*)take((size_t)MM * 512 * 2);    // 32 MiB
  __bf16* h1       = (__bf16*)take((size_t)MM * 512 * 2);    // 32 MiB
  __bf16* summed   = (__bf16*)take((size_t)MM * DDICT * 2);  // 16 MiB
  float*  emis     = (float*) take((size_t)MM * 3 * 4);
  __bf16* l0Wih_b  = (__bf16*)take((size_t)NGATE * DIN0 * 2);
  __bf16* l0Whh_b  = (__bf16*)take((size_t)NGATE * HH * 2);
  __bf16* l1Wih_b  = (__bf16*)take((size_t)NGATE * 512 * 2);
  __bf16* l1Whh_b  = (__bf16*)take((size_t)NGATE * HH * 2);
  __bf16* dW_b     = (__bf16*)take((size_t)DDICT * DDICT * 2);

  auto cvt = [&](const float* s, __bf16* d, int n) {
    f32_to_bf16_k<<<(n + 255) / 256, 256, 0, stream>>>(s, d, n);
  };
  cvt(l0Wih, l0Wih_b, NGATE * DIN0);
  cvt(l0Whh, l0Whh_b, NGATE * HH);
  cvt(l1Wih, l1Wih_b, NGATE * 512);
  cvt(l1Whh, l1Whh_b, NGATE * HH);
  cvt(dW,    dW_b,    DDICT * DDICT);

  { int n = MM * DBERT;
    pack_seq_k<<<(n + 255) / 256, 256, 0, stream>>>(seq, combined, n); }

  { int n = MM * 64;
    dict_gather_k<<<(n + 255) / 256, 256, 0, stream>>>(didx, dval, demb, summed); }

  // dict linear + relu -> combined cols [768,1024) as bf16
  {
    dim3 grid(MM / 64, DDICT / 256);
    gemm_wmma_k<<<grid, 256, 0, stream>>>(summed, DDICT, dW_b, DDICT, db,
                                          nullptr, combined + DBERT, DIN0,
                                          DDICT, GF_RELU | GF_BF16);
  }
  // L0 input projection: pre = combined @ [Wih0;Wih1]^T + bias  (f32)
  {
    dim3 grid(MM / 64, NGATE / 256);
    gemm_wmma_k<<<grid, 256, 0, stream>>>(combined, DIN0, l0Wih_b, DIN0, l0b,
                                          pre, nullptr, NGATE, DIN0, 0);
  }
  size_t smem = (size_t)BB * HH * 2 + (size_t)BB * G4H * 4;  // 288 KB
  (void)hipFuncSetAttribute((const void*)lstm_rec_k,
                            hipFuncAttributeMaxDynamicSharedMemorySize, (int)smem);
  lstm_rec_k<<<2, 1024, smem, stream>>>(l0Whh_b, pre, h0);

  // L1 input projection (reuse pre buffer): pre = h0 @ l1Wih^T + bias
  {
    dim3 grid(MM / 64, NGATE / 256);
    gemm_wmma_k<<<grid, 256, 0, stream>>>(h0, 512, l1Wih_b, 512, l1b,
                                          pre, nullptr, NGATE, 512, 0);
  }
  lstm_rec_k<<<2, 1024, smem, stream>>>(l1Whh_b, pre, h1);

  emissions_k<<<MM / 256, 256, 0, stream>>>(h1, posW, posb, emis);

  crf_k<<<1, BB, 0, stream>>>(emis, plabel, amask, crfS, crfE, crfT,
                              (float*)d_out);
  (void)in_sizes; (void)n_in; (void)out_size; (void)ws_size;
}
